// SetAbstractionLayer_57775900066181
// MI455X (gfx1250) — matrix operand order repaired and checked by reading, not compiled, not run on hardware
//
#include <hip/hip_runtime.h>

// ---------------------------------------------------------------------------
// PointNet++ SetAbstraction for MI455X (gfx1250, wave32, WMMA bf16)
//   B=8, N=8192, NPOINT=2048, NSAMPLE=32, D_FEAT=64, MLP=[64,64,128]
// ---------------------------------------------------------------------------

#define B_       8
#define N_       8192
#define NPOINT_  2048
#define NSAMPLE_ 32
#define DFEAT_   64
#define NCOLS    (B_ * NPOINT_ * NSAMPLE_)   // 524288 = 2^19
#define LOG2NCOLS 19
#define R2_      0.04f
#define EPS_     1e-5f

typedef __attribute__((ext_vector_type(16))) __bf16 v16bf;
typedef __attribute__((ext_vector_type(8)))  __bf16 v8bf;
typedef __attribute__((ext_vector_type(8)))  float  v8f;

static __device__ __forceinline__ unsigned short f2bf_bits(float f) {
  __bf16 h = (__bf16)f;
  return __builtin_bit_cast(unsigned short, h);
}
static __device__ __forceinline__ float bf2f(unsigned short u) {
  unsigned int x = ((unsigned int)u) << 16;
  return __builtin_bit_cast(float, x);
}

// ---------------------------------------------------------------------------
// 1) Farthest point sampling: one workgroup per batch, xyz in LDS,
//    per-thread distances in registers, wave32 shfl argmax reduction.
//    Writes new_xyz directly into d_out (first output).
// ---------------------------------------------------------------------------
__global__ __launch_bounds__(1024) void fps_kernel(
    const float* __restrict__ xyz, float* __restrict__ out_xyz) {
  extern __shared__ char smem[];
  float* lx   = (float*)smem;
  float* ly   = lx + N_;
  float* lz   = ly + N_;
  float* rval = lz + N_;            // [32]
  int*   ridx = (int*)(rval + 32);  // [32]
  int*   sfar = ridx + 32;          // [1]

  const int b    = blockIdx.x;
  const int tid  = threadIdx.x;
  const int lane = tid & 31;
  const int wid  = tid >> 5;

  const float* xb = xyz + (size_t)b * N_ * 3;
  for (int j = tid; j < N_; j += 1024) {
    lx[j] = xb[j * 3 + 0];
    ly[j] = xb[j * 3 + 1];
    lz[j] = xb[j * 3 + 2];
  }
  float dist[8];
#pragma unroll
  for (int t = 0; t < 8; ++t) dist[t] = 1e10f;
  __syncthreads();

  int far = 0;
  for (int i = 0; i < NPOINT_; ++i) {
    if (tid == 0) {
      float* o = out_xyz + ((size_t)b * NPOINT_ + i) * 3;
      o[0] = lx[far]; o[1] = ly[far]; o[2] = lz[far];
    }
    const float cx = lx[far], cy = ly[far], cz = lz[far];
    float bv = -1.0f; int bi = 0;
#pragma unroll
    for (int t = 0; t < 8; ++t) {
      const int j = tid + t * 1024;
      const float dx = lx[j] - cx, dy = ly[j] - cy, dz = lz[j] - cz;
      float d = dx * dx + dy * dy + dz * dz;
      d = fminf(dist[t], d);
      dist[t] = d;
      if (d > bv || (d == bv && j < bi)) { bv = d; bi = j; }
    }
#pragma unroll
    for (int m = 16; m >= 1; m >>= 1) {
      const float ov = __shfl_xor(bv, m, 32);
      const int   oi = __shfl_xor(bi, m, 32);
      if (ov > bv || (ov == bv && oi < bi)) { bv = ov; bi = oi; }
    }
    if (lane == 0) { rval[wid] = bv; ridx[wid] = bi; }
    __syncthreads();
    if (wid == 0) {
      bv = rval[lane]; bi = ridx[lane];
#pragma unroll
      for (int m = 16; m >= 1; m >>= 1) {
        const float ov = __shfl_xor(bv, m, 32);
        const int   oi = __shfl_xor(bi, m, 32);
        if (ov > bv || (ov == bv && oi < bi)) { bv = ov; bi = oi; }
      }
      if (lane == 0) *sfar = bi;
    }
    __syncthreads();
    far = *sfar;
    __syncthreads();  // protect rval/ridx/sfar reuse next iteration
  }
}

// ---------------------------------------------------------------------------
// 2) Ball query + gather + concat -> bf16 activation matrix X0[96][NCOLS].
//    One wave per center (8 centers / 256-thread block, all same batch);
//    xyz in LDS; ordered first-32 selection via ballot + prefix popcount.
// ---------------------------------------------------------------------------
__global__ __launch_bounds__(256) void ballgroup_kernel(
    const float* __restrict__ xyz, const float* __restrict__ points,
    const float* __restrict__ new_xyz, unsigned short* __restrict__ X0) {
  extern __shared__ char smem[];
  float* lx = (float*)smem;
  float* ly = lx + N_;
  float* lz = ly + N_;
  int* ssel = (int*)(lz + N_);  // [8][32]

  const int tid  = threadIdx.x;
  const int lane = tid & 31;
  const int wid  = tid >> 5;
  const int gc   = blockIdx.x * 8 + wid;          // global center index
  const int b    = blockIdx.x / (NPOINT_ / 8);    // batch of this block

  const float* xb = xyz + (size_t)b * N_ * 3;
  for (int j = tid; j < N_; j += 256) {
    lx[j] = xb[j * 3 + 0];
    ly[j] = xb[j * 3 + 1];
    lz[j] = xb[j * 3 + 2];
  }
  __syncthreads();

  const float cx = new_xyz[(size_t)gc * 3 + 0];
  const float cy = new_xyz[(size_t)gc * 3 + 1];
  const float cz = new_xyz[(size_t)gc * 3 + 2];

  int cnt = 0;
  for (int j0 = 0; j0 < N_ && cnt < NSAMPLE_; j0 += 32) {
    const int j = j0 + lane;
    const float dx = lx[j] - cx, dy = ly[j] - cy, dz = lz[j] - cz;
    const float d2 = dx * dx + dy * dy + dz * dz;
    const unsigned int mask = (unsigned int)__ballot(d2 <= R2_);
    const int pos = cnt + __popc(mask & ((1u << lane) - 1u));
    if (((mask >> lane) & 1u) && pos < NSAMPLE_) ssel[wid * 32 + pos] = j;
    cnt += __popc(mask);
  }
  if (cnt == 0 && lane == 0) ssel[wid * 32] = 0;  // safeguard (unreachable)

  const int myidx = ssel[wid * 32 + ((lane < cnt) ? lane : 0)];
  const int col   = gc * NSAMPLE_ + lane;

  // channels 0..2: grouped_xyz - new_xyz
  X0[(size_t)0 * NCOLS + col] = f2bf_bits(lx[myidx] - cx);
  X0[(size_t)1 * NCOLS + col] = f2bf_bits(ly[myidx] - cy);
  X0[(size_t)2 * NCOLS + col] = f2bf_bits(lz[myidx] - cz);
  // channels 3..66: gathered point features
  const float* pb = points + ((size_t)b * N_ + myidx) * DFEAT_;
#pragma unroll 4
  for (int d = 0; d < DFEAT_; ++d)
    X0[(size_t)(3 + d) * NCOLS + col] = f2bf_bits(pb[d]);
  // zero K-padding rows 67..95
  for (int k = 3 + DFEAT_; k < 96; ++k)
    X0[(size_t)k * NCOLS + col] = (unsigned short)0;
}

// ---------------------------------------------------------------------------
// 3) GEMM Y = W @ X + b with v_wmma_f32_16x16x32_bf16.
//    Block: 256 threads = 8 waves, 256-column tile. Each wave owns TWO
//    16-column stripes (cols w*16 and 128+w*16) so every A (weight) fragment
//    feeds two WMMAs. W and the X tile staged in LDS (padded stride 104
//    halves: 16B-aligned rows, spread banks). Epilogue folds both stripes'
//    partial sums before the cross-lane BN reduction (shfl) + atomics.
// ---------------------------------------------------------------------------
template <int COUT, int CIN, int KPAD>
__global__ __launch_bounds__(256) void gemm_bn_kernel(
    const unsigned short* __restrict__ X, const float* __restrict__ W,
    const float* __restrict__ bias, unsigned short* __restrict__ Y,
    float* __restrict__ stats) {
  constexpr int SW = 104;  // LDS stride (halves), 16B aligned rows
  constexpr int SX = 104;
  extern __shared__ char smem[];
  unsigned short* sW = (unsigned short*)smem;  // [COUT][SW]
  unsigned short* sX = sW + COUT * SW;         // [256][SX] (col-major tile)

  const int tid    = threadIdx.x;
  const int lane   = tid & 31;
  const int wid    = tid >> 5;
  const int laneLo = lane & 15;
  const int laneHi = lane >> 4;

  // stage weights (f32 -> bf16, zero-pad K to KPAD)
  for (int i = tid; i < COUT * KPAD; i += 256) {
    const int m = i / KPAD, k = i - m * KPAD;
    const float v = (k < CIN) ? W[m * CIN + k] : 0.0f;
    sW[m * SW + k] = f2bf_bits(v);
  }
  // stage 256-column activation tile, transposed to [col][k]
  const size_t c0 = (size_t)blockIdx.x * 256;
  for (int i = tid; i < KPAD * 256; i += 256) {
    const int k = i >> 8, col = i & 255;
    sX[col * SX + k] = X[((size_t)k << LOG2NCOLS) + c0 + col];
  }
  __syncthreads();

  const int colw = wid * 16 + laneLo;  // stripe 0 column; stripe 1 = +128

#pragma unroll
  for (int mt = 0; mt < COUT / 16; ++mt) {
    v8f acc0 = {0.f, 0.f, 0.f, 0.f, 0.f, 0.f, 0.f, 0.f};
    v8f acc1 = {0.f, 0.f, 0.f, 0.f, 0.f, 0.f, 0.f, 0.f};
#pragma unroll
    for (int kt = 0; kt < KPAD / 32; ++kt) {
      // A fragment (16x32 bf16): lanes 0-15 -> K base 0, lanes 16-31 -> +16;
      // halves 0..15 = K[base..base+15]
      union { v16bf v; v8bf h[2]; } a;
      const int arow  = mt * 16 + laneLo;
      const int abase = arow * SW + kt * 32 + laneHi * 16;
      a.h[0] = *(const v8bf*)(sW + abase);
      a.h[1] = *(const v8bf*)(sW + abase + 8);
      // B fragments (32x16 bf16): lane -> N=lane%16, K = laneHi*16 + h
      union { v16bf v; v8bf h[2]; } b0, b1;
      const int bbase0 = colw * SX + kt * 32 + laneHi * 16;
      b0.h[0] = *(const v8bf*)(sX + bbase0);
      b0.h[1] = *(const v8bf*)(sX + bbase0 + 8);
      const int bbase1 = (colw + 128) * SX + kt * 32 + laneHi * 16;
      b1.h[0] = *(const v8bf*)(sX + bbase1);
      b1.h[1] = *(const v8bf*)(sX + bbase1 + 8);
      acc0 = __builtin_amdgcn_wmma_f32_16x16x32_bf16(
          false, a.v, false, b0.v, (short)0, acc0, false, false);
      acc1 = __builtin_amdgcn_wmma_f32_16x16x32_bf16(
          false, a.v, false, b1.v, (short)0, acc1, false, false);
    }
    // epilogue: bias add, bf16 stores, BN partial sums (both stripes folded)
#pragma unroll
    for (int r = 0; r < 8; ++r) {
      const int m = mt * 16 + r + laneHi * 8;  // C layout: VGPR r -> M=r / M=r+8
      const float bm = bias[m];
      const float y0 = acc0[r] + bm;
      const float y1 = acc1[r] + bm;
      Y[((size_t)m << LOG2NCOLS) + c0 + colw]       = f2bf_bits(y0);
      Y[((size_t)m << LOG2NCOLS) + c0 + colw + 128] = f2bf_bits(y1);
      float s = y0 + y1, sq = y0 * y0 + y1 * y1;
#pragma unroll
      for (int msk = 8; msk >= 1; msk >>= 1) {
        s  += __shfl_xor(s, msk, 32);
        sq += __shfl_xor(sq, msk, 32);
      }
      if (laneLo == 0) {
        atomicAdd(&stats[m], s);
        atomicAdd(&stats[COUT + m], sq);
      }
    }
  }
}

// ---------------------------------------------------------------------------
// 4) BN finalize: stats -> per-channel scale/shift
// ---------------------------------------------------------------------------
__global__ void bn_finalize_kernel(const float* __restrict__ stats,
                                   const float* __restrict__ gamma,
                                   const float* __restrict__ beta,
                                   float* __restrict__ scsh, int cout) {
  const int c = blockIdx.x * blockDim.x + threadIdx.x;
  if (c < cout) {
    const float inv  = 1.0f / (float)NCOLS;
    const float mean = stats[c] * inv;
    const float var  = stats[cout + c] * inv - mean * mean;
    const float sc   = gamma[c] * rsqrtf(var + EPS_);
    scsh[c]        = sc;
    scsh[cout + c] = beta[c] - mean * sc;
  }
}

// 5) normalize + relu -> next layer bf16 input
__global__ __launch_bounds__(256) void normrelu_kernel(
    const unsigned short* __restrict__ Y, const float* __restrict__ scsh,
    unsigned short* __restrict__ Xn, int cout) {
  const size_t i = (size_t)blockIdx.x * 256 + threadIdx.x;
  if (i < ((size_t)cout << LOG2NCOLS)) {
    const int ch = (int)(i >> LOG2NCOLS);
    const float v = fmaxf(0.f, bf2f(Y[i]) * scsh[ch] + scsh[cout + ch]);
    Xn[i] = f2bf_bits(v);
  }
}

// 6) fused BN+relu+max over NSAMPLE for the last layer -> new_points.
//    32 contiguous bf16 samples read as four uint4 (global_load_b128).
__global__ __launch_bounds__(256) void maxpool_kernel(
    const unsigned short* __restrict__ Y, const float* __restrict__ scsh,
    float* __restrict__ out_points) {
  const int i = blockIdx.x * 256 + threadIdx.x;
  if (i < (B_ * NPOINT_) * 128) {
    const int gc = i & (B_ * NPOINT_ - 1);
    const int ch = i >> 14;
    const uint4* p4 =
        (const uint4*)(Y + ((size_t)ch << LOG2NCOLS) + (size_t)gc * NSAMPLE_);
    const float sc = scsh[ch], sh = scsh[128 + ch];
    float m = 0.f;  // relu output >= 0
#pragma unroll
    for (int q = 0; q < 4; ++q) {
      const uint4 v = p4[q];
      const unsigned int w[4] = {v.x, v.y, v.z, v.w};
#pragma unroll
      for (int e = 0; e < 4; ++e) {
        const float f0 = bf2f((unsigned short)(w[e] & 0xffffu));
        const float f1 = bf2f((unsigned short)(w[e] >> 16));
        m = fmaxf(m, fmaxf(0.f, f0 * sc + sh));
        m = fmaxf(m, fmaxf(0.f, f1 * sc + sh));
      }
    }
    out_points[(size_t)gc * 128 + ch] = m;
  }
}

__global__ void zero_kernel(float* __restrict__ p, int n) {
  const int i = blockIdx.x * blockDim.x + threadIdx.x;
  if (i < n) p[i] = 0.f;
}

// ---------------------------------------------------------------------------
extern "C" void kernel_launch(void* const* d_in, const int* in_sizes, int n_in,
                              void* d_out, int out_size, void* d_ws,
                              size_t ws_size, hipStream_t stream) {
  const float* xyz    = (const float*)d_in[0];
  const float* points = (const float*)d_in[1];
  const float* W1 = (const float*)d_in[2];
  const float* b1 = (const float*)d_in[3];
  const float* g1 = (const float*)d_in[4];
  const float* e1 = (const float*)d_in[5];
  const float* W2 = (const float*)d_in[6];
  const float* b2 = (const float*)d_in[7];
  const float* g2 = (const float*)d_in[8];
  const float* e2 = (const float*)d_in[9];
  const float* W3 = (const float*)d_in[10];
  const float* b3 = (const float*)d_in[11];
  const float* g3 = (const float*)d_in[12];
  const float* e3 = (const float*)d_in[13];

  float* out        = (float*)d_out;
  float* out_xyz    = out;                             // (8,2048,3)
  float* out_points = out + (size_t)B_ * NPOINT_ * 3;  // (8,2048,128)

  // workspace layout (bf16 ping-pong buffers + BN stats)
  unsigned short* X0 = (unsigned short*)d_ws;        // 96  x NCOLS bf16
  unsigned short* Yb = X0 + (size_t)96 * NCOLS;      // 128 x NCOLS bf16
  float* stats = (float*)(Yb + (size_t)128 * NCOLS); // 256 f32
  float* scsh  = stats + 256;                        // 256 f32

  // 1) FPS -> new_xyz (into d_out)
  {
    const size_t lds = (size_t)3 * N_ * 4 + 32 * 4 + 32 * 4 + 16;
    fps_kernel<<<B_, 1024, lds, stream>>>(xyz, out_xyz);
  }
  // 2) ball query + gather + concat -> X0 (bf16, K padded to 96)
  {
    const size_t lds = (size_t)3 * N_ * 4 + 8 * 32 * 4;
    ballgroup_kernel<<<NPOINT_, 256, lds, stream>>>(xyz, points, out_xyz, X0);
  }

  const int gemm_grid = NCOLS / 256;  // 2048
  const size_t lds64  = (size_t)(64 * 104 + 256 * 104) * 2;
  const size_t lds128 = (size_t)(128 * 104 + 256 * 104) * 2;
  const unsigned int nr_grid = (unsigned int)(((size_t)64 << LOG2NCOLS) / 256);

  // Layer 1: 67 -> 64
  zero_kernel<<<1, 256, 0, stream>>>(stats, 256);
  gemm_bn_kernel<64, 67, 96><<<gemm_grid, 256, lds64, stream>>>(X0, W1, b1, Yb, stats);
  bn_finalize_kernel<<<1, 64, 0, stream>>>(stats, g1, e1, scsh, 64);
  normrelu_kernel<<<nr_grid, 256, 0, stream>>>(Yb, scsh, X0, 64);

  // Layer 2: 64 -> 64
  zero_kernel<<<1, 256, 0, stream>>>(stats, 256);
  gemm_bn_kernel<64, 64, 64><<<gemm_grid, 256, lds64, stream>>>(X0, W2, b2, Yb, stats);
  bn_finalize_kernel<<<1, 64, 0, stream>>>(stats, g2, e2, scsh, 64);
  normrelu_kernel<<<nr_grid, 256, 0, stream>>>(Yb, scsh, X0, 64);

  // Layer 3: 64 -> 128, BN+relu+max fused into pooling
  zero_kernel<<<1, 256, 0, stream>>>(stats, 256);
  gemm_bn_kernel<128, 64, 64><<<gemm_grid, 256, lds128, stream>>>(X0, W3, b3, Yb, stats);
  bn_finalize_kernel<<<1, 128, 0, stream>>>(stats, g3, e3, scsh, 128);
  maxpool_kernel<<<(B_ * NPOINT_ * 128) / 256, 256, 0, stream>>>(Yb, scsh, out_points);
}
